// peepLSTM_73512660238909
// MI455X (gfx1250) — matrix operand order repaired and checked by reading, not compiled, not run on hardware
//
#include <hip/hip_runtime.h>
#include <hip/hip_bf16.h>
#include <stdint.h>

// Problem constants (match the reference)
#define H_ 1024
#define B_ 128
#define T_ 256
#define E_ 64
#define C_ 10
#define NT_ 32          // output columns per workgroup

typedef __attribute__((ext_vector_type(16))) __bf16 v16bf;
typedef __attribute__((ext_vector_type(8)))  __bf16 v8bf;
typedef __attribute__((ext_vector_type(8)))  float  v8f;
typedef int v4i __attribute__((vector_size(16)));   // matches async-LDS builtin pointee

#define GLOBAL_AS __attribute__((address_space(1)))
#define LDS_AS    __attribute__((address_space(3)))

#if defined(__has_builtin)
#if __has_builtin(__builtin_amdgcn_global_load_async_to_lds_b128)
#define HAVE_ASYNC_LDS 1
#endif
#if __has_builtin(__builtin_amdgcn_s_wait_asynccnt)
#define HAVE_WAIT_ASYNC 1
#endif
#endif

__device__ __forceinline__ float sigmoidf_(float v) {
    return 1.0f / (1.0f + __expf(-v));
}

__device__ __forceinline__ void wait_async_zero() {
#if defined(HAVE_ASYNC_LDS)
#if defined(HAVE_WAIT_ASYNC)
    __builtin_amdgcn_s_wait_asynccnt(0);
#else
    asm volatile("s_wait_asynccnt 0" ::: "memory");
#endif
#endif
}

// Stage one B-slab chunk set: lbuf[g][nn][kk] = wt[g][n0+nn][k0+kk]
// Slab is 3 gates x NT_ cols x 32 K = 3072 bf16; threads 0..191 move 16 each.
__device__ __forceinline__ void stage_slab(const __bf16* __restrict__ wt,
                                           __bf16* lbuf, int tid, int n0, int k0) {
    if (tid < 192) {
#pragma unroll
        for (int cch = 0; cch < 2; ++cch) {
            int e  = tid * 16 + cch * 8;
            int kk = e & 31;
            int nn = (e >> 5) & (NT_ - 1);
            int g  = e >> 10;                       // NT_*32 == 1024
            const __bf16* src = wt + (size_t)g * H_ * H_ + (size_t)(n0 + nn) * H_ + (k0 + kk);
            __bf16* dst = lbuf + (g * NT_ + nn) * 32 + kk;
#if defined(HAVE_ASYNC_LDS)
            // global -> LDS without touching VGPRs; tracked by ASYNCcnt
            __builtin_amdgcn_global_load_async_to_lds_b128(
                (GLOBAL_AS v4i*)(uintptr_t)src,
                (LDS_AS v4i*)(unsigned)(uintptr_t)dst,
                0, 0);
#else
            *(v8bf*)dst = *(const v8bf*)src;
#endif
        }
    }
}

// ---------------------------------------------------------------------------
// Convert + transpose the three recurrent weight matrices to bf16, n-major:
//   wt[g][n][k] = bf16( W_g[k][n] ),  g in {f,i,o}
// ---------------------------------------------------------------------------
__global__ void prep_wt_kernel(const float* __restrict__ Wfh,
                               const float* __restrict__ Wih,
                               const float* __restrict__ Woh,
                               __bf16* __restrict__ wt) {
    unsigned idx = blockIdx.x * 256u + threadIdx.x;   // over 3 * H * H
    unsigned g = idx >> 20;                           // H*H == 2^20
    unsigned r = idx & ((H_ * H_) - 1);
    unsigned n = r >> 10;
    unsigned k = r & (H_ - 1);
    const float* W = (g == 0) ? Wfh : ((g == 1) ? Wih : Woh);
    wt[idx] = (__bf16)W[(size_t)k * H_ + n];
}

// ---------------------------------------------------------------------------
// Gate tables: tab[g][r][h] = emb[r] . W_gx[:,h] + b_g[h]   (g=3: sigmoid)
// ---------------------------------------------------------------------------
__global__ void prep_tables_kernel(const float* __restrict__ emb,
                                   const float* __restrict__ Wfx, const float* __restrict__ bfv,
                                   const float* __restrict__ Wix, const float* __restrict__ biv,
                                   const float* __restrict__ Wox, const float* __restrict__ bov,
                                   const float* __restrict__ Wcx, const float* __restrict__ bcv,
                                   float* __restrict__ tab) {
    int idx = blockIdx.x * 256 + threadIdx.x;         // over 4 * 3 * H
    if (idx >= 4 * 3 * H_) return;
    int g = idx / (3 * H_);
    int r = (idx / H_) % 3;
    int h = idx % H_;
    const float* W;
    const float* bias;
    if      (g == 0) { W = Wfx; bias = bfv; }
    else if (g == 1) { W = Wix; bias = biv; }
    else if (g == 2) { W = Wox; bias = bov; }
    else             { W = Wcx; bias = bcv; }
    float s = bias[h];
#pragma unroll 8
    for (int e = 0; e < E_; ++e) s += emb[r * E_ + e] * W[e * H_ + h];
    if (g == 3) s = sigmoidf_(s);
    tab[idx] = s;
}

__global__ void zero_state_kernel(float* __restrict__ cF,
                                  __bf16* __restrict__ c0,
                                  __bf16* __restrict__ c1) {
    int i = blockIdx.x * 256 + threadIdx.x;
    if (i < B_ * H_) {
        cF[i] = 0.0f;
        c0[i] = (__bf16)0.0f;
        c1[i] = (__bf16)0.0f;
    }
}

// ---------------------------------------------------------------------------
// One LSTM timestep.
//   grid  : H/NT_ = 32 workgroups, each owns output columns [n0, n0+NT_)
//   block : 256 threads = 8 waves, wave w owns rows [w*16, w*16+16)
// S_g = c @ W_gh for g in {f,i,o} via v_wmma_f32_16x16x32_bf16 with
// double-buffered, async-staged LDS B-slabs; then the peephole cell update.
// At t == T-1 also writes h = tanh(c_new) * o.
// ---------------------------------------------------------------------------
__global__ __launch_bounds__(256)
void lstm_step_kernel(const __bf16* __restrict__ wt,    // [3][H][H] n-major bf16
                      const float*  __restrict__ tab,   // [4][3][H]
                      const int*    __restrict__ x,     // [B][T]
                      const __bf16* __restrict__ cSrc,  // [B][H] bf16 (read)
                      __bf16*       __restrict__ cDst,  // [B][H] bf16 (write)
                      float*        __restrict__ cF,    // [B][H] f32 cell
                      float*        __restrict__ hOut,  // [B][H] (written at t==T-1)
                      int t) {
    __shared__ __bf16 lb[2][3 * NT_ * 32];              // 2 x 6 KB double buffer

    const int tid  = threadIdx.x;
    const int wave = tid >> 5;
    const int lane = tid & 31;
    const int n0   = blockIdx.x * NT_;
    const int m0   = wave * 16;
    const int lm   = lane & 15;      // row (A) / col (B) within 16
    const int kh   = lane >> 4;      // K-half select

    const v8f vzero = {0.f, 0.f, 0.f, 0.f, 0.f, 0.f, 0.f, 0.f};
    v8f acc[3][2];
#pragma unroll
    for (int g = 0; g < 3; ++g)
#pragma unroll
        for (int j = 0; j < 2; ++j) acc[g][j] = vzero;

    // prologue: stage first slab
    stage_slab(wt, &lb[0][0], tid, n0, 0);
    wait_async_zero();
    __syncthreads();

    int p = 0;
    for (int k0 = 0; k0 < H_; k0 += 32) {
        // issue async stage of the NEXT slab into the other buffer
        if (k0 + 32 < H_) stage_slab(wt, &lb[p ^ 1][0], tid, n0, k0 + 32);

        // ---- A fragment: rows m0..m0+15 of cSrc, K chunk [k0, k0+32)
        const __bf16* ap = cSrc + (size_t)(m0 + lm) * H_ + k0;
        v8bf alo = *(const v8bf*)(ap + 8 * kh);
        v8bf ahi = *(const v8bf*)(ap + 16 + 8 * kh);
        v16bf a = __builtin_shufflevector(alo, ahi,
                                          0, 1, 2, 3, 4, 5, 6, 7,
                                          8, 9, 10, 11, 12, 13, 14, 15);
        __builtin_prefetch(ap + 32, 0, 3);

        // ---- 6 WMMAs: 3 gates x 2 N-subtiles from current LDS buffer
        const __bf16* lbuf = &lb[p][0];
#pragma unroll
        for (int g = 0; g < 3; ++g) {
#pragma unroll
            for (int j = 0; j < 2; ++j) {
                const __bf16* bp = lbuf + (g * NT_ + j * 16 + lm) * 32 + 8 * kh;
                v8bf blo = *(const v8bf*)bp;
                v8bf bhi = *(const v8bf*)(bp + 16);
                v16bf b = __builtin_shufflevector(blo, bhi,
                                                  0, 1, 2, 3, 4, 5, 6, 7,
                                                  8, 9, 10, 11, 12, 13, 14, 15);
                acc[g][j] = __builtin_amdgcn_wmma_f32_16x16x32_bf16(
                    false, a, false, b, (short)0, acc[g][j], false, false);
            }
        }

        // next slab staged + all waves done reading current buffer
        wait_async_zero();
        __syncthreads();
        p ^= 1;
    }

    // ---- epilogue: peephole gate update for this wave's 16xNT_ tile.
    // acc[g][j][r] lives at row m = m0 + r + 8*kh, col n = n0 + j*16 + lm.
    const float* Tf = tab + 0 * 3 * H_;
    const float* Ti = tab + 1 * 3 * H_;
    const float* To = tab + 2 * 3 * H_;
    const float* Tc = tab + 3 * 3 * H_;

#pragma unroll
    for (int r = 0; r < 8; ++r) {
        const int m  = m0 + r + 8 * kh;
        const int xi = x[m * T_ + t];
#pragma unroll
        for (int j = 0; j < 2; ++j) {
            const int n = n0 + j * 16 + lm;
            float f  = sigmoidf_(Tf[xi * H_ + n] + acc[0][j][r]);
            float ii = sigmoidf_(Ti[xi * H_ + n] + acc[1][j][r]);
            float o  = sigmoidf_(To[xi * H_ + n] + acc[2][j][r]);
            float gc = Tc[xi * H_ + n];
            size_t off = (size_t)m * H_ + n;
            float cnew = gc * ii + cF[off] * f;
            cF[off]   = cnew;
            cDst[off] = (__bf16)cnew;
            if (t == T_ - 1) hOut[off] = tanhf(cnew) * o;
        }
    }
}

// ---------------------------------------------------------------------------
// Head: p = h_T @ W_ph + b ; out = log_softmax(p)
// ---------------------------------------------------------------------------
__global__ void head_kernel(const float* __restrict__ hbuf,
                            const float* __restrict__ Wph,
                            const float* __restrict__ Wpb,
                            float* __restrict__ out) {
    int b = blockIdx.x * blockDim.x + threadIdx.x;
    if (b >= B_) return;
    float p[C_];
#pragma unroll
    for (int c = 0; c < C_; ++c) p[c] = Wpb[c];
    for (int h = 0; h < H_; ++h) {
        float hv = hbuf[(size_t)b * H_ + h];
#pragma unroll
        for (int c = 0; c < C_; ++c) p[c] += hv * Wph[h * C_ + c];
    }
    float mx = p[0];
#pragma unroll
    for (int c = 1; c < C_; ++c) mx = fmaxf(mx, p[c]);
    float s = 0.0f;
#pragma unroll
    for (int c = 0; c < C_; ++c) s += __expf(p[c] - mx);
    float lse = __logf(s) + mx;
#pragma unroll
    for (int c = 0; c < C_; ++c) out[b * C_ + c] = p[c] - lse;
}

// ---------------------------------------------------------------------------
extern "C" void kernel_launch(void* const* d_in, const int* in_sizes, int n_in,
                              void* d_out, int out_size, void* d_ws, size_t ws_size,
                              hipStream_t stream) {
    const int*   x   = (const int*)  d_in[0];
    const float* emb = (const float*)d_in[1];
    const float* Wfx = (const float*)d_in[2];
    const float* Wfh = (const float*)d_in[3];
    const float* Wfb = (const float*)d_in[4];
    const float* Wix = (const float*)d_in[5];
    const float* Wih = (const float*)d_in[6];
    const float* Wib = (const float*)d_in[7];
    const float* Wox = (const float*)d_in[8];
    const float* Woh = (const float*)d_in[9];
    const float* Wob = (const float*)d_in[10];
    const float* Wcx = (const float*)d_in[11];
    const float* Wcb = (const float*)d_in[12];
    const float* Wph = (const float*)d_in[13];
    const float* Wpb = (const float*)d_in[14];
    float* out = (float*)d_out;

    // Workspace carve (~7.9 MB total)
    char* ws = (char*)d_ws;
    size_t off = 0;
    __bf16* wt  = (__bf16*)(ws + off); off += (size_t)3 * H_ * H_ * sizeof(__bf16); // 6 MB
    float*  tab = (float*) (ws + off); off += (size_t)4 * 3 * H_ * sizeof(float);   // 48 KB
    float*  cF  = (float*) (ws + off); off += (size_t)B_ * H_ * sizeof(float);      // 512 KB
    __bf16* c0  = (__bf16*)(ws + off); off += (size_t)B_ * H_ * sizeof(__bf16);     // 256 KB
    __bf16* c1  = (__bf16*)(ws + off); off += (size_t)B_ * H_ * sizeof(__bf16);     // 256 KB
    float*  hb  = (float*) (ws + off); off += (size_t)B_ * H_ * sizeof(float);      // 512 KB
    (void)ws_size; (void)in_sizes; (void)n_in; (void)out_size;

    prep_wt_kernel    <<<(3 * H_ * H_) / 256, 256, 0, stream>>>(Wfh, Wih, Woh, wt);
    prep_tables_kernel<<<(4 * 3 * H_ + 255) / 256, 256, 0, stream>>>(
        emb, Wfx, Wfb, Wix, Wib, Wox, Wob, Wcx, Wcb, tab);
    zero_state_kernel <<<(B_ * H_) / 256, 256, 0, stream>>>(cF, c0, c1);

    __bf16* src = c0;
    __bf16* dst = c1;
    for (int t = 0; t < T_; ++t) {
        lstm_step_kernel<<<H_ / NT_, 256, 0, stream>>>(wt, tab, x, src, dst, cF, hb, t);
        __bf16* tmp = src; src = dst; dst = tmp;
    }

    head_kernel<<<2, 64, 0, stream>>>(hb, Wph, Wpb, out);
}
